// TEMPMEBlock_12266426598095
// MI455X (gfx1250) — matrix operand order repaired and checked by reading, not compiled, not run on hardware
//
#include <hip/hip_runtime.h>
#include <hip/hip_bf16.h>
#include <math.h>

// ---------------- CDNA5 WMMA types ----------------
typedef __attribute__((ext_vector_type(8)))  __bf16 v8bf;
typedef __attribute__((ext_vector_type(16))) __bf16 v16bf;
typedef __attribute__((ext_vector_type(8)))  float  v8f;

#define EPS 1e-8f
#define NTOK 1024
#define DIM  512
#define KIMG 512
#define KINTRA 256
#define NCB  64   // C*B
#define NPAIR 48  // (C-1)*B

// workspace layout (bytes)
#define OFS_IDX   0                         // int   [64][512]
#define OFS_INVN  (OFS_IDX  + 64*512*4)     // float [64][512]  (1/||row||)
#define OFS_TOP   (OFS_INVN + 64*512*4)     // int   [48][512]
#define OFS_PROC  (OFS_TOP  + 48*512*4)     // bf16  [64][512][512]

__device__ __forceinline__ float wred_add(float v) {
    #pragma unroll
    for (int off = 16; off > 0; off >>= 1) v += __shfl_xor(v, off, 32);
    return v;
}

// =====================================================================
// K0: scores -> full bitonic sort (desc, tie: lower idx first, matching
//     jax.lax.top_k) -> gather top-512 rows to bf16 + fp32 inv-norms.
// grid: 64 blocks (one per clip-batch), 1024 threads (32 waves).
// =====================================================================
__global__ __launch_bounds__(1024)
void k_select_gather(const float* __restrict__ clips,
                     const float* __restrict__ W,
                     const float* __restrict__ bias,
                     int* __restrict__ idxOut,
                     float* __restrict__ invNorms,
                     __hip_bfloat16* __restrict__ procB)
{
    __shared__ float sW[DIM];
    __shared__ float sVal[NTOK];
    __shared__ int   sIdx[NTOK];

    const int tid  = threadIdx.x;
    const int lane = tid & 31;
    const int wave = tid >> 5;
    const int cb   = blockIdx.x;
    const float bv = bias[0];
    const float* xb = clips + (size_t)cb * NTOK * DIM;

    if (tid < DIM) sW[tid] = W[tid];
    __syncthreads();

    // ---- scores: one wave per row, lane-strided dot ----
    #pragma unroll 1
    for (int i = 0; i < 32; ++i) {
        const int row = wave * 32 + i;
        const float* p = xb + (size_t)row * DIM;
        float s = 0.f;
        #pragma unroll
        for (int e = 0; e < 16; ++e) { int d = lane + 32 * e; s += p[d] * sW[d]; }
        s = wred_add(s);
        if (lane == 0) { sVal[row] = s + bv; sIdx[row] = row; }
    }
    __syncthreads();

    // ---- bitonic sort 1024 pairs, descending value, ascending idx on ties ----
    for (int k = 2; k <= NTOK; k <<= 1) {
        for (int j = k >> 1; j > 0; j >>= 1) {
            const int i = tid, ixj = i ^ j;
            if (ixj > i) {
                const bool dirDesc = ((i & k) == 0);
                float v1 = sVal[i], v2 = sVal[ixj];
                int   i1 = sIdx[i], i2 = sIdx[ixj];
                // "i should come after ixj" in descending order
                const bool after = (v1 < v2) || (v1 == v2 && i1 > i2);
                if (after == dirDesc) {
                    sVal[i] = v2; sVal[ixj] = v1;
                    sIdx[i] = i2; sIdx[ixj] = i1;
                }
            }
            __syncthreads();
        }
    }

    if (tid < KIMG) idxOut[(size_t)cb * KIMG + tid] = sIdx[tid];

    // ---- gather + bf16 convert + reciprocal norms ----
    #pragma unroll 1
    for (int i = 0; i < 16; ++i) {
        const int k = wave * 16 + i;                 // 32 waves * 16 = 512
        const int src = sIdx[k];
        const float* p = xb + (size_t)src * DIM;
        __hip_bfloat16* q = procB + ((size_t)cb * KIMG + k) * DIM;
        float ss = 0.f;
        #pragma unroll
        for (int e = 0; e < 16; ++e) {
            int d = lane + 32 * e;
            float v = p[d];
            q[d] = __float2bfloat16(v);
            ss += v * v;
        }
        ss = wred_add(ss);
        if (lane == 0) invNorms[(size_t)cb * KIMG + k] = rsqrtf(ss);  // v_rsq_f32
    }
}

// =====================================================================
// K1: cross-clip cosine-sim GEMM via v_wmma_f32_16x16x32_bf16 + fused
//     per-row argmax. grid: (32 row-tiles, 48 pairs), 256 threads.
// Per wave: 16 rows x 64 cols (4 independent 16x16 accumulators),
// K=512; A fragment loaded once per k-step and reused by 4 WMMAs.
// =====================================================================
__global__ __launch_bounds__(256)
void k_cross_argmax(const unsigned short* __restrict__ procRaw,
                    const float* __restrict__ invn,
                    int* __restrict__ top)
{
    __shared__ float candV[16][8];
    __shared__ int   candI[16][8];

    const int m    = blockIdx.y;          // pair: x1 = proc[m], x2 = proc[m+16]
    const int n0   = blockIdx.x * 16;     // row tile of x1
    const int lane = threadIdx.x & 31;
    const int w    = threadIdx.x >> 5;    // 8 waves
    const int lh   = lane & 15;
    const bool hiHalf = (lane >= 16);

    const __bf16* A = reinterpret_cast<const __bf16*>(procRaw) + (size_t)m * KIMG * DIM;
    const __bf16* B = reinterpret_cast<const __bf16*>(procRaw) + (size_t)(m + 16) * KIMG * DIM;
    const float* inv1 = invn + (size_t)m * KIMG;
    const float* inv2 = invn + (size_t)(m + 16) * KIMG;

    // A-frag lane row/K offsets (16x32), contiguous k order used here, same as B:
    //  lanes 0-15:  K = k0..k0+15
    //  lanes 16-31: K = k0+16..k0+31
    const int arow  = n0 + lh;
    const int kaOff = hiHalf ? 16 : 0;
    // B-frag (32x16): lanes 0-15 hold K=k0..k0+15 of column lh; lanes 16-31 K=k0+16..k0+31
    const int kbOff = hiHalf ? 16 : 0;

    float rowInv1[8];
    #pragma unroll
    for (int r = 0; r < 8; ++r) rowInv1[r] = inv1[n0 + r + (hiHalf ? 8 : 0)];

    int jq[4];
    const __bf16* pb0[4];
    #pragma unroll
    for (int q = 0; q < 4; ++q) {
        jq[q]  = (w + 8 * q) * 16 + lh;           // ascending in q
        pb0[q] = B + (size_t)jq[q] * DIM + kbOff;
    }

    v8f acc[4];
    #pragma unroll
    for (int q = 0; q < 4; ++q) acc[q] = (v8f){0.f,0.f,0.f,0.f,0.f,0.f,0.f,0.f};

    const v8bf* pa0 = reinterpret_cast<const v8bf*>(A + (size_t)arow * DIM + kaOff);

    // K loop: load A once, feed 4 independent accumulators (4x fewer A loads,
    // back-to-back independent WMMAs).
    #pragma unroll 2
    for (int kt = 0; kt < 16; ++kt) {             // K = 512 / 32
        const v8bf* pa = pa0 + (size_t)kt * 4;    // +32 bf16 per step
        v8bf alo = pa[0];
        v8bf ahi = pa[1];                          // +8 elements
        v16bf afrag = __builtin_shufflevector(alo, ahi,
            0, 1, 2, 3, 4, 5, 6, 7, 8, 9, 10, 11, 12, 13, 14, 15);
        #pragma unroll
        for (int q = 0; q < 4; ++q) {
            v16bf bfrag = *reinterpret_cast<const v16bf*>(pb0[q] + (size_t)kt * 32);
            acc[q] = __builtin_amdgcn_wmma_f32_16x16x32_bf16(
                false, afrag, false, bfrag, (short)0, acc[q], false, false);
        }
    }

    // ---- fused argmax epilogue: multiplies only, no divides ----
    float bestv[8]; int besti[8];
    #pragma unroll
    for (int r = 0; r < 8; ++r) { bestv[r] = -3.4e38f; besti[r] = 0; }

    #pragma unroll
    for (int q = 0; q < 4; ++q) {                 // ascending j -> first occurrence
        const float i2 = inv2[jq[q]];
        #pragma unroll
        for (int r = 0; r < 8; ++r) {
            // C layout: vgpr r, lanes 0-15 -> M=r, lanes 16-31 -> M=r+8; N = lane&15
            float val = acc[q][r] * rowInv1[r] * i2;
            if (val > bestv[r]) { bestv[r] = val; besti[r] = jq[q]; }
        }
    }

    // reduce across the 16 columns held by each 16-lane half (tie: smaller j)
    #pragma unroll
    for (int r = 0; r < 8; ++r) {
        float v = bestv[r]; int i = besti[r];
        #pragma unroll
        for (int off = 8; off > 0; off >>= 1) {
            float ov = __shfl_xor(v, off, 16);
            int   oi = __shfl_xor(i, off, 16);
            if (ov > v || (ov == v && oi < i)) { v = ov; i = oi; }
        }
        if (lh == 0) {
            const int rr = r + (hiHalf ? 8 : 0);
            candV[rr][w] = v; candI[rr][w] = i;
        }
    }
    __syncthreads();

    if (threadIdx.x < 16) {
        float v = candV[threadIdx.x][0]; int i = candI[threadIdx.x][0];
        #pragma unroll
        for (int ww = 1; ww < 8; ++ww) {
            float ov = candV[threadIdx.x][ww]; int oi = candI[threadIdx.x][ww];
            if (ov > v || (ov == v && oi < i)) { v = ov; i = oi; }
        }
        top[(size_t)m * KIMG + n0 + threadIdx.x] = i;
    }
}

// =====================================================================
// K2: merged rows recomputed fp32 from original clips; cosine vs row 0;
//     bitonic top-256 set; LDS ds_add_f32 accumulation; mean -> out.
// grid: 48 blocks, 512 threads (16 waves).
// =====================================================================
__global__ __launch_bounds__(512)
void k_intra(const float* __restrict__ clips,
             const int* __restrict__ idxArr,
             const int* __restrict__ top,
             float* __restrict__ out)
{
    __shared__ float r0[DIM];
    __shared__ float cosV[KIMG];
    __shared__ int   cosI[KIMG];
    __shared__ float sum[DIM];
    __shared__ float sNm0;

    const int m    = blockIdx.x;
    const int tid  = threadIdx.x;
    const int lane = tid & 31;
    const int w    = tid >> 5;            // 16 waves
    const int cb1  = m;                   // m = c*16 + b  -> x1 clip-batch
    const int cb2  = m + 16;              // x2 clip-batch
    const float* clip1 = clips + (size_t)cb1 * NTOK * DIM;
    const float* clip2 = clips + (size_t)cb2 * NTOK * DIM;
    const int* id1 = idxArr + (size_t)cb1 * KIMG;
    const int* id2 = idxArr + (size_t)cb2 * KIMG;
    const int* tp  = top + (size_t)m * KIMG;

    // ---- merged row 0, fp32 ----
    {
        const int t0 = tp[0];
        const float* p1 = clip1 + (size_t)id1[t0] * DIM;
        const float* p2 = clip2 + (size_t)id2[t0] * DIM;
        r0[tid] = 0.5f * (p1[tid] + p2[tid]);
        sum[tid] = 0.f;
    }
    __syncthreads();
    if (w == 0) {
        float ss = 0.f;
        #pragma unroll
        for (int e = 0; e < 16; ++e) { float v = r0[lane + 32 * e]; ss += v * v; }
        ss = wred_add(ss);
        if (lane == 0) sNm0 = sqrtf(ss);
    }
    __syncthreads();

    // ---- cosine(merged[0], merged[n]) for all n ----
    #pragma unroll 1
    for (int i = 0; i < 32; ++i) {
        const int n = w + 16 * i;
        const int t = tp[n];
        const float* p1 = clip1 + (size_t)id1[t] * DIM;
        const float* p2 = clip2 + (size_t)id2[t] * DIM;
        float dot = 0.f, ss = 0.f;
        #pragma unroll
        for (int e = 0; e < 16; ++e) {
            int d = lane + 32 * e;
            float mv = 0.5f * (p1[d] + p2[d]);
            dot += mv * r0[d];
            ss  += mv * mv;
        }
        dot = wred_add(dot);
        ss  = wred_add(ss);
        if (lane == 0) {
            cosV[n] = dot / fmaxf(sNm0 * sqrtf(ss), EPS);
            cosI[n] = n;
        }
    }
    __syncthreads();

    // ---- bitonic sort 512 (desc, tie: lower idx) ----
    for (int k = 2; k <= KIMG; k <<= 1) {
        for (int j = k >> 1; j > 0; j >>= 1) {
            const int i = tid, ixj = i ^ j;
            if (ixj > i) {
                const bool dirDesc = ((i & k) == 0);
                float v1 = cosV[i], v2 = cosV[ixj];
                int   i1 = cosI[i], i2 = cosI[ixj];
                const bool after = (v1 < v2) || (v1 == v2 && i1 > i2);
                if (after == dirDesc) {
                    cosV[i] = v2; cosV[ixj] = v1;
                    cosI[i] = i2; cosI[ixj] = i1;
                }
            }
            __syncthreads();
        }
    }

    // ---- mean of the 256 selected merged rows ----
    float acc[16];
    #pragma unroll
    for (int e = 0; e < 16; ++e) acc[e] = 0.f;
    #pragma unroll 1
    for (int l = 0; l < 16; ++l) {
        const int q = w * 16 + l;            // 16 waves * 16 = 256 selected rows
        const int n = cosI[q];
        const int t = tp[n];
        const float* p1 = clip1 + (size_t)id1[t] * DIM;
        const float* p2 = clip2 + (size_t)id2[t] * DIM;
        #pragma unroll
        for (int e = 0; e < 16; ++e) {
            int d = lane + 32 * e;
            acc[e] += 0.5f * (p1[d] + p2[d]);
        }
    }
    #pragma unroll
    for (int e = 0; e < 16; ++e) atomicAdd(&sum[lane + 32 * e], acc[e]);  // ds_add_f32
    __syncthreads();

    out[(size_t)m * DIM + tid] = sum[tid] * (1.0f / (float)KINTRA);
}

// =====================================================================
extern "C" void kernel_launch(void* const* d_in, const int* in_sizes, int n_in,
                              void* d_out, int out_size, void* d_ws, size_t ws_size,
                              hipStream_t stream) {
    const float* clips = (const float*)d_in[0];   // (4,16,1024,512) f32
    const float* W     = (const float*)d_in[1];   // (512,1) f32
    const float* bias  = (const float*)d_in[2];   // (1,) f32
    float* out = (float*)d_out;                   // (3,16,1,512) f32

    char* ws = (char*)d_ws;                       // needs ~34 MB
    int*            idxArr = (int*)  (ws + OFS_IDX);
    float*          invN   = (float*)(ws + OFS_INVN);
    int*            top    = (int*)  (ws + OFS_TOP);
    __hip_bfloat16* procB  = (__hip_bfloat16*)(ws + OFS_PROC);

    k_select_gather<<<NCB, 1024, 0, stream>>>(clips, W, bias, idxArr, invN, procB);
    k_cross_argmax<<<dim3(32, NPAIR), 256, 0, stream>>>(
        (const unsigned short*)procB, invN, top);
    k_intra<<<NPAIR, 512, 0, stream>>>(clips, idxArr, top, out);
}
